// Encoder_ATTENTION_33818572489008
// MI455X (gfx1250) — compile-verified, matrix operand order checked by reading, not run on hardware
//
#include <hip/hip_runtime.h>

typedef __attribute__((ext_vector_type(16))) __bf16 v16bf;
typedef __attribute__((ext_vector_type(8)))  float  v8f;

#define B_TOT   8192
#define NEI     64
#define DIM     128
#define D2      256
#define NBLK    1024
#define WSTRIDE 264   // ushorts per LDS row: 256 data + 8 pad (528 B, 16B-aligned, odd 16B-granule)

union FragU {
    uint4 q[2];
    v16bf v;
};

__device__ __forceinline__ unsigned int pack_bf16x2(float a, float b) {
    unsigned int ua = __float_as_uint(a);
    unsigned int ub = __float_as_uint(b);
    ua += 0x7FFFu + ((ua >> 16) & 1u);   // round-to-nearest-even
    ub += 0x7FFFu + ((ub >> 16) & 1u);
    return (ua >> 16) | (ub & 0xFFFF0000u);
}

__device__ __forceinline__ float fast_tanh(float x) {
    float y;
    asm("v_tanh_f32 %0, %1" : "=v"(y) : "v"(x));
    return y;
}

// ---- kernel 1: attn_W_w (f32, 256x256 row-major) -> bf16 in workspace ----
__global__ void convert_w_bf16(const float* __restrict__ w,
                               unsigned short* __restrict__ o, int n) {
    int i = blockIdx.x * blockDim.x + threadIdx.x;
    if (i < n) {
        unsigned int u = __float_as_uint(w[i]);
        u += 0x7FFFu + ((u >> 16) & 1u);
        o[i] = (unsigned short)(u >> 16);
    }
}

// ---- kernel 2: fused gather/project/GEMM/tanh/softmax with async HBM->LDS pipeline ----
__launch_bounds__(256, 1)
__global__ void attn_fused(const float* __restrict__ e_emb,
                           const float* __restrict__ rw,
                           const float* __restrict__ w_r_table,
                           const float* __restrict__ zq_table,
                           const unsigned short* __restrict__ Wbf,
                           const float* __restrict__ Wbias,
                           const float* __restrict__ Ua,
                           const float* __restrict__ Ub,
                           const int*   __restrict__ nei_rid,
                           const int*   __restrict__ q_rid,
                           float* __restrict__ out)
{
    __shared__ unsigned short W_lds[D2  * WSTRIDE];  // 135168 B  (bf16 W, row o = output)
    __shared__ unsigned short A_lds[NEI * WSTRIDE];  //  33792 B  (bf16 [z_q | e_proj])
    __shared__ float E_stage[2][NEI * DIM];          //  65536 B  (double-buffered raw e rows)
    __shared__ float s_bias[D2];
    __shared__ float s_u[D2];
    __shared__ float s_alpha[NEI];

    const int tid  = threadIdx.x;
    const int lane = tid & 31;
    const int wv   = tid >> 5;      // wave 0..7
    const int hi   = lane >> 4;     // half-wave (K sub-chunk select)
    const int l16  = lane & 15;

    // ---- stage bf16 W into LDS once (coalesced uint4) ----
    for (int i = tid; i < (D2 * D2) / 8; i += 256) {   // 8192 x 16B
        int row = i >> 5;
        int c   = i & 31;
        uint4 d = ((const uint4*)Wbf)[i];
        *(uint4*)&W_lds[row * WSTRIDE + c * 8] = d;
    }
    if (tid < D2) { s_bias[tid] = Wbias[tid]; s_u[tid] = Ua[tid]; }
    const float ub = Ub[0];

    // ---- async prefetch of this wave's 8 e-rows for the FIRST b into buffer 0 ----
    // Each global_load_async_to_lds_b128: 32 lanes x 16B = 512B = one full e row.
    {
        const unsigned long long gbase =
            (unsigned long long)(uintptr_t)(e_emb + (size_t)blockIdx.x * NEI * DIM);
        const unsigned lbase = (unsigned)(uintptr_t)(&E_stage[0][0]);  // flat low32 == LDS offset
        #pragma unroll
        for (int j = 0; j < 8; ++j) {
            unsigned voff = (unsigned)(((wv * 8 + j) * DIM) * 4 + lane * 16);
            unsigned laddr = lbase + voff;
            asm volatile("global_load_async_to_lds_b128 %0, %1, %2"
                         :: "v"(laddr), "v"(voff), "s"(gbase) : "memory");
        }
    }
    __syncthreads();

    const int nt0 = wv * 2;   // this wave: n-tiles {nt0, nt0+1}, all 4 m-tiles

    int cur = 0;
    for (int b = blockIdx.x; b < B_TOT; b += NBLK, cur ^= 1) {
        if (tid < NEI) s_alpha[tid] = 0.0f;

        // ---- issue async prefetch for NEXT b into the other buffer ----
        {
            const int bn = (b + NBLK < B_TOT) ? (b + NBLK) : b;   // last iter: dead re-fetch
            const unsigned long long gbase =
                (unsigned long long)(uintptr_t)(e_emb + (size_t)bn * NEI * DIM);
            const unsigned lbase = (unsigned)(uintptr_t)(&E_stage[cur ^ 1][0]);
            #pragma unroll
            for (int j = 0; j < 8; ++j) {
                unsigned voff = (unsigned)(((wv * 8 + j) * DIM) * 4 + lane * 16);
                unsigned laddr = lbase + voff;
                asm volatile("global_load_async_to_lds_b128 %0, %1, %2"
                             :: "v"(laddr), "v"(voff), "s"(gbase) : "memory");
            }
        }
        // retire the OLD prefetch (<=8 outstanding leaves only the 8 just issued;
        // async loads complete in order, and this wave reads only rows it loaded itself)
        asm volatile("s_wait_asynccnt 0x8" ::: "memory");

        // z_q row -> bf16 (4 values/lane), shared by all 64 neighbor rows
        const int qr = q_rid[b];
        float4 zq = ((const float4*)(zq_table + (size_t)qr * DIM))[lane];
        uint2 zqb;
        zqb.x = pack_bf16x2(zq.x, zq.y);
        zqb.y = pack_bf16x2(zq.z, zq.w);

        // ---- build A: hyperplane projection e - ((e.w)/(w.w)) w, one row per wave-iter ----
        #pragma unroll
        for (int i = 0; i < 8; ++i) {
            const int n   = wv * 8 + i;
            const int rid = nei_rid[b * NEI + n];
            float4 wr = ((const float4*)(w_r_table + (size_t)rid * DIM))[lane];
            float4 ev = ((const float4*)&E_stage[cur][n * DIM])[lane];
            float s1 = ev.x*wr.x + ev.y*wr.y + ev.z*wr.z + ev.w*wr.w;
            float s2 = wr.x*wr.x + wr.y*wr.y + wr.z*wr.z + wr.w*wr.w;
            #pragma unroll
            for (int o = 16; o > 0; o >>= 1) {
                s1 += __shfl_xor(s1, o, 32);
                s2 += __shfl_xor(s2, o, 32);
            }
            const float coef = s1 / s2;
            uint2 pb;
            pb.x = pack_bf16x2(ev.x - coef*wr.x, ev.y - coef*wr.y);
            pb.y = pack_bf16x2(ev.z - coef*wr.z, ev.w - coef*wr.w);
            *(uint2*)&A_lds[n * WSTRIDE + lane * 4]       = zqb;  // concat[0:128] = z_q
            *(uint2*)&A_lds[n * WSTRIDE + DIM + lane * 4] = pb;   // concat[128:256] = e_proj
        }
        __syncthreads();

        // ---- GEMM: H = A(64x256) . W^T(256x256), bf16 WMMA, f32 accumulate ----
        v8f acc[4][2];
        #pragma unroll
        for (int m = 0; m < 4; ++m)
            #pragma unroll
            for (int n2 = 0; n2 < 2; ++n2)
                acc[m][n2] = (v8f)(0.0f);

        #pragma unroll
        for (int k = 0; k < 8; ++k) {
            const int kb = k * 32 + hi * 8;   // per-lane K base (ISA 16-bit A layout)
            FragU a[4];
            #pragma unroll
            for (int m = 0; m < 4; ++m) {
                const unsigned short* p = &A_lds[(m * 16 + l16) * WSTRIDE + kb];
                a[m].q[0] = *(const uint4*)p;          // K: kb..kb+7
                a[m].q[1] = *(const uint4*)(p + 16);   // K: kb+16..kb+23
            }
            #pragma unroll
            for (int n2 = 0; n2 < 2; ++n2) {
                FragU bb;
                const unsigned short* p = &W_lds[((nt0 + n2) * 16 + l16) * WSTRIDE + kb];
                bb.q[0] = *(const uint4*)p;
                bb.q[1] = *(const uint4*)(p + 16);
                #pragma unroll
                for (int m = 0; m < 4; ++m) {
                    acc[m][n2] = __builtin_amdgcn_wmma_f32_16x16x32_bf16(
                        false, a[m].v, false, bb.v, (short)0, acc[m][n2], false, false);
                }
            }
        }

        // ---- epilogue: alpha[row] += sum_o tanh(h+bias)*u  (C layout: VGPR r -> M=r+8*hi, N=l16) ----
        #pragma unroll
        for (int n2 = 0; n2 < 2; ++n2) {
            const int   o  = (nt0 + n2) * 16 + l16;
            const float bs = s_bias[o];
            const float uu = s_u[o];
            #pragma unroll
            for (int m = 0; m < 4; ++m) {
                #pragma unroll
                for (int r = 0; r < 8; ++r) {
                    float hv = fast_tanh(acc[m][n2][r] + bs) * uu;
                    atomicAdd(&s_alpha[m * 16 + hi * 8 + r], hv);
                }
            }
        }
        __syncthreads();

        // ---- softmax over 64 neighbors + add rw (single wave) ----
        if (tid < 32) {
            float v0 = s_alpha[tid]      + ub;
            float v1 = s_alpha[tid + 32] + ub;
            float mx = fmaxf(v0, v1);
            #pragma unroll
            for (int o = 16; o > 0; o >>= 1) mx = fmaxf(mx, __shfl_xor(mx, o, 32));
            float e0 = __expf(v0 - mx);
            float e1 = __expf(v1 - mx);
            float s  = e0 + e1;
            #pragma unroll
            for (int o = 16; o > 0; o >>= 1) s += __shfl_xor(s, o, 32);
            const float inv = 1.0f / s;
            out[(size_t)b * NEI + tid]      = e0 * inv + rw[(size_t)b * NEI + tid];
            out[(size_t)b * NEI + tid + 32] = e1 * inv + rw[(size_t)b * NEI + tid + 32];
        }
        __syncthreads();
    }
}

extern "C" void kernel_launch(void* const* d_in, const int* in_sizes, int n_in,
                              void* d_out, int out_size, void* d_ws, size_t ws_size,
                              hipStream_t stream) {
    const float* e_emb = (const float*)d_in[0];
    const float* rwp   = (const float*)d_in[1];
    const float* wrt   = (const float*)d_in[2];
    const float* zqt   = (const float*)d_in[3];
    const float* Ww    = (const float*)d_in[4];
    const float* Wb    = (const float*)d_in[5];
    const float* Ua    = (const float*)d_in[6];
    const float* Ub    = (const float*)d_in[7];
    const int*   nrid  = (const int*)d_in[8];
    const int*   qrid  = (const int*)d_in[9];
    unsigned short* Wbf = (unsigned short*)d_ws;   // 131072 B of scratch

    convert_w_bf16<<<(D2 * D2 + 255) / 256, 256, 0, stream>>>(Ww, Wbf, D2 * D2);
    attn_fused<<<NBLK, 256, 0, stream>>>(e_emb, rwp, wrt, zqt, Wbf, Wb, Ua, Ub,
                                         nrid, qrid, (float*)d_out);
}
